// SplitMemoryGraphTransformer_35742717837916
// MI455X (gfx1250) — compile-verified, matrix operand order checked by reading, not run on hardware
//
#include <hip/hip_runtime.h>
#include <hip/hip_bf16.h>

typedef __attribute__((ext_vector_type(16))) __bf16 v16bf;
typedef __attribute__((ext_vector_type(8)))  __bf16 v8bf;
typedef __attribute__((ext_vector_type(8)))  float  v8f;

#define N_NODES  100000
#define D_MODEL  256
#define N_SLOTS  512
#define BLOCK_THREADS 256
#define WAVES 8
#define ROWS_PER_BLOCK (WAVES * 16)
#define SLOT_TILE 32
#define N_ITERS (N_SLOTS / SLOT_TILE)   // 16

// LDS strides (in bf16 elements); all 16B fragment chunks stay 16B-aligned
#define MT_STRIDE  264   // 32 x 264  (row = slot, cols = d)   528B/row = 33*16
#define MTT_STRIDE 48    // 256 x 48  (row = d,   cols = slot)  96B/row =  6*16
#define P_STRIDE   48    // per-wave 16 x 48                    96B/row

// workspace layout (bytes)
#define WD_BYTES (N_SLOTS * D_MODEL * 2)        // 256KB per mem, d-major bf16
#define WS_BYTES (D_MODEL * N_SLOTS * 2)        // 256KB per mem, slot-major bf16

__device__ __forceinline__ float half_max(float v) {
    v = fmaxf(v, __shfl_xor(v, 1, 32));
    v = fmaxf(v, __shfl_xor(v, 2, 32));
    v = fmaxf(v, __shfl_xor(v, 4, 32));
    v = fmaxf(v, __shfl_xor(v, 8, 32));
    return v;
}
__device__ __forceinline__ float half_sum(float v) {
    v += __shfl_xor(v, 1, 32);
    v += __shfl_xor(v, 2, 32);
    v += __shfl_xor(v, 4, 32);
    v += __shfl_xor(v, 8, 32);
    return v;
}

// Build a 16-element bf16 fragment from two contiguous 16B LDS chunks.
__device__ __forceinline__ v16bf ld16(const __bf16* p0, const __bf16* p1) {
    v8bf lo = *(const v8bf*)p0;
    v8bf hi = *(const v8bf*)p1;
    v16bf r;
#pragma unroll
    for (int e = 0; e < 8; ++e) { r[e] = lo[e]; r[8 + e] = hi[e]; }
    return r;
}

// 16B async DMA: global -> LDS (ASYNCcnt-tracked, no VGPR data round-trip).
// LDS flat-pointer low 32 bits == workgroup-relative LDS byte address (ISA 10.2).
__device__ __forceinline__ void async_ld16B(unsigned lds_addr, const void* gaddr) {
    asm volatile("global_load_async_to_lds_b128 %0, %1, off"
                 :: "v"(lds_addr), "v"((unsigned long long)(uintptr_t)gaddr)
                 : "memory");
}

// ---------------- prep kernel: fp32 mem -> bf16 in two orientations ----------------
__global__ __launch_bounds__(256)
void convert_mem_bf16(const float* __restrict__ mem_attr,
                      const float* __restrict__ mem_struct,
                      unsigned short* __restrict__ wd_a, unsigned short* __restrict__ wd_s,
                      unsigned short* __restrict__ ws_a, unsigned short* __restrict__ ws_s)
{
    const float* mem = (blockIdx.y == 0) ? mem_attr : mem_struct;
    __bf16* wd = (__bf16*)((blockIdx.y == 0) ? wd_a : wd_s);
    __bf16* ws = (__bf16*)((blockIdx.y == 0) ? ws_a : ws_s);
    const int slot = blockIdx.x;        // 0..511
    const int d    = threadIdx.x;       // 0..255
    __bf16 b = (__bf16)mem[slot * D_MODEL + d];
    wd[slot * D_MODEL + d] = b;         // d-major   (score B operand)
    ws[d * N_SLOTS + slot] = b;         // slot-major (O-update B operand)
}

// issue one 32x256 bf16 tile (both orientations) as 8 async B128 chunks per thread
__device__ __forceinline__ void stage_async(const __bf16* __restrict__ wd,
                                            const __bf16* __restrict__ ws,
                                            int sbase, __bf16* mt, __bf16* mtt, int tid)
{
    // Mt: 32 rows x 512B payload -> 1024 chunks of 16B
#pragma unroll
    for (int u = 0; u < 4; ++u) {
        const int c   = tid + u * 256;
        const int row = c >> 5;          // 0..31 (slot)
        const int col = c & 31;          // 16B unit within row
        unsigned lds  = (unsigned)(uintptr_t)(mt + row * MT_STRIDE) + col * 16;
        const char* g = (const char*)(wd + (size_t)(sbase + row) * D_MODEL) + col * 16;
        async_ld16B(lds, g);
    }
    // MtT: 256 rows x 64B payload -> 1024 chunks of 16B
#pragma unroll
    for (int u = 0; u < 4; ++u) {
        const int c   = tid + u * 256;
        const int row = c >> 2;          // 0..255 (d)
        const int col = c & 3;
        unsigned lds  = (unsigned)(uintptr_t)(mtt + row * MTT_STRIDE) + col * 16;
        const char* g = (const char*)(ws + (size_t)row * N_SLOTS + sbase) + col * 16;
        async_ld16B(lds, g);
    }
}

// ---------------- main fused kernel ----------------
__global__ __launch_bounds__(BLOCK_THREADS)
void fused_mem_read(const float* __restrict__ q,
                    const unsigned short* __restrict__ wd_a, const unsigned short* __restrict__ wd_s,
                    const unsigned short* __restrict__ ws_a, const unsigned short* __restrict__ ws_s,
                    float* __restrict__ out)
{
    __shared__ __align__(16) __bf16 MtBuf [2][SLOT_TILE * MT_STRIDE];
    __shared__ __align__(16) __bf16 MtTBuf[2][D_MODEL  * MTT_STRIDE];
    __shared__ __align__(16) __bf16 Pb    [WAVES * 16 * P_STRIDE];

    const __bf16* wd = (const __bf16*)((blockIdx.y == 0) ? wd_a : wd_s);
    const __bf16* ws = (const __bf16*)((blockIdx.y == 0) ? ws_a : ws_s);
    float* outb = out + (size_t)blockIdx.y * (size_t)N_NODES * (2 * D_MODEL);

    const int tid  = threadIdx.x;
    const int lane = tid & 31;
    const int wid  = tid >> 5;
    const int h    = lane >> 4;   // half-wave select
    const int ln   = lane & 15;   // row (A) / column (B,C,D) index
    const int rb   = blockIdx.x * ROWS_PER_BLOCK + wid * 16;

    // ---- preload this wave's 16x256 Q tile as bf16 A-fragments (ISA 16-bit A layout) ----
    int arow = rb + ln; if (arow > N_NODES - 1) arow = N_NODES - 1;
    const float* qrow = q + (size_t)arow * D_MODEL;
    v16bf QA[8];
#pragma unroll
    for (int c = 0; c < 8; ++c) {
        const float* pk0 = qrow + c * 32 + 8 * h;        // K = 32c + 8h + e      (e=0..7)
        const float* pk1 = qrow + c * 32 + 16 + 8 * h;   // K = 32c + 16 + 8h + e
        float4 a0 = *(const float4*)(pk0);
        float4 a1 = *(const float4*)(pk0 + 4);
        float4 b0 = *(const float4*)(pk1);
        float4 b1 = *(const float4*)(pk1 + 4);
        QA[c][0]  = (__bf16)a0.x; QA[c][1]  = (__bf16)a0.y; QA[c][2]  = (__bf16)a0.z; QA[c][3]  = (__bf16)a0.w;
        QA[c][4]  = (__bf16)a1.x; QA[c][5]  = (__bf16)a1.y; QA[c][6]  = (__bf16)a1.z; QA[c][7]  = (__bf16)a1.w;
        QA[c][8]  = (__bf16)b0.x; QA[c][9]  = (__bf16)b0.y; QA[c][10] = (__bf16)b0.z; QA[c][11] = (__bf16)b0.w;
        QA[c][12] = (__bf16)b1.x; QA[c][13] = (__bf16)b1.y; QA[c][14] = (__bf16)b1.z; QA[c][15] = (__bf16)b1.w;
    }

    v8f O[16];
#pragma unroll
    for (int t = 0; t < 16; ++t)
#pragma unroll
        for (int i = 0; i < 8; ++i) O[t][i] = 0.0f;

    float m_run[8], l_run[8];
#pragma unroll
    for (int i = 0; i < 8; ++i) { m_run[i] = -1e30f; l_run[i] = 0.0f; }

    __bf16* pwave = Pb + wid * 16 * P_STRIDE;

    // ---- double-buffered async pipeline over 16 slot-tiles of 32 ----
    stage_async(wd, ws, 0, MtBuf[0], MtTBuf[0], tid);
    int buf = 0;
    for (int it = 0; it < N_ITERS; ++it) {
        if (it + 1 < N_ITERS) {
            stage_async(wd, ws, (it + 1) * SLOT_TILE, MtBuf[buf ^ 1], MtTBuf[buf ^ 1], tid);
            // async loads retire in order: <=8 outstanding => current tile landed
            asm volatile("s_wait_asynccnt 0x8" ::: "memory");
        } else {
            asm volatile("s_wait_asynccnt 0x0" ::: "memory");
        }
        __syncthreads();   // all threads' current-tile chunks visible

        const __bf16* Mt  = MtBuf[buf];
        const __bf16* MtT = MtTBuf[buf];

        // ---- scores: S(16x32) = Qtile(16x256) @ Mtile^T, two 16x16 D-tiles ----
        v8f S0, S1;
#pragma unroll
        for (int i = 0; i < 8; ++i) { S0[i] = 0.f; S1[i] = 0.f; }
#pragma unroll
        for (int c = 0; c < 8; ++c) {
            const __bf16* r0 = Mt + (0 * 16 + ln) * MT_STRIDE + c * 32 + 8 * h;
            v16bf B0 = ld16(r0, r0 + 16);
            S0 = __builtin_amdgcn_wmma_f32_16x16x32_bf16(false, QA[c], false, B0, (short)0, S0, false, false);
            const __bf16* r1 = Mt + (1 * 16 + ln) * MT_STRIDE + c * 32 + 8 * h;
            v16bf B1 = ld16(r1, r1 + 16);
            S1 = __builtin_amdgcn_wmma_f32_16x16x32_bf16(false, QA[c], false, B1, (short)0, S1, false, false);
        }

        // ---- online softmax over this 32-slot tile (exp in place) ----
        float f[8];
#pragma unroll
        for (int i = 0; i < 8; ++i) {
            float rmax = half_max(fmaxf(S0[i], S1[i]));   // rows live in 16-lane halves
            float mnew = fmaxf(m_run[i], rmax);
            f[i]  = __expf(m_run[i] - mnew);
            S0[i] = __expf(S0[i] - mnew);
            S1[i] = __expf(S1[i] - mnew);
            float rs = half_sum(S0[i] + S1[i]);
            l_run[i] = l_run[i] * f[i] + rs;
            m_run[i] = mnew;
        }
#pragma unroll
        for (int t = 0; t < 16; ++t)
#pragma unroll
            for (int i = 0; i < 8; ++i) O[t][i] *= f[i];

        // ---- P: D-layout -> per-wave LDS -> A-layout bf16 fragment ----
#pragma unroll
        for (int i = 0; i < 8; ++i) {
            pwave[(8 * h + i) * P_STRIDE + ln]      = (__bf16)S0[i];
            pwave[(8 * h + i) * P_STRIDE + 16 + ln] = (__bf16)S1[i];
        }
        asm volatile("s_wait_dscnt 0" ::: "memory");   // wave-local LDS RAW
        const __bf16* pr = pwave + ln * P_STRIDE + 8 * h;
        v16bf PA = ld16(pr, pr + 16);

        // ---- O(16x256) += P(16x32) @ Mtile(32x256): B from slot-major MtT ----
#pragma unroll
        for (int t = 0; t < 16; ++t) {
            const __bf16* br = MtT + (16 * t + ln) * MTT_STRIDE + 8 * h;
            v16bf Bt = ld16(br, br + 16);
            O[t] = __builtin_amdgcn_wmma_f32_16x16x32_bf16(false, PA, false, Bt, (short)0, O[t], false, false);
        }

        __syncthreads();   // all waves done reading buf before it is refilled
        buf ^= 1;
    }

    // ---- epilogue: normalize + store read_memory; copy query passthrough ----
    float rinv[8];
#pragma unroll
    for (int i = 0; i < 8; ++i) rinv[i] = 1.0f / l_run[i];

#pragma unroll
    for (int i = 0; i < 8; ++i) {
        const int row = rb + 8 * h + i;
        if (row < N_NODES) {
            float* orow = outb + (size_t)row * (2 * D_MODEL) + D_MODEL;
#pragma unroll
            for (int t = 0; t < 16; ++t)
                orow[16 * t + ln] = O[t][i] * rinv[i];
        }
    }

    for (int r = 0; r < 16; ++r) {
        const int row = rb + r;
        if (row >= N_NODES) break;
        const float4* s = (const float4*)(q + (size_t)row * D_MODEL);
        float4*       d = (float4*)(outb + (size_t)row * (2 * D_MODEL));
        d[lane]      = s[lane];
        d[lane + 32] = s[lane + 32];
    }
}

extern "C" void kernel_launch(void* const* d_in, const int* in_sizes, int n_in,
                              void* d_out, int out_size, void* d_ws, size_t ws_size,
                              hipStream_t stream) {
    (void)in_sizes; (void)n_in; (void)out_size; (void)ws_size;
    const float* q  = (const float*)d_in[0];
    const float* ma = (const float*)d_in[1];
    const float* ms = (const float*)d_in[2];
    float* out = (float*)d_out;

    char* wsb = (char*)d_ws;
    unsigned short* wd_a = (unsigned short*)(wsb);
    unsigned short* wd_s = (unsigned short*)(wsb + WD_BYTES);
    unsigned short* ws_a = (unsigned short*)(wsb + 2 * WD_BYTES);
    unsigned short* ws_s = (unsigned short*)(wsb + 2 * WD_BYTES + WS_BYTES);

    convert_mem_bf16<<<dim3(N_SLOTS, 2, 1), 256, 0, stream>>>(ma, ms, wd_a, wd_s, ws_a, ws_s);

    dim3 grid((N_NODES + ROWS_PER_BLOCK - 1) / ROWS_PER_BLOCK, 2, 1);
    fused_mem_read<<<grid, BLOCK_THREADS, 0, stream>>>(q, wd_a, wd_s, ws_a, ws_s, out);
}
